// MultiscaleLinearAttention_14542759264948
// MI455X (gfx1250) — matrix operand order repaired and checked by reading, not compiled, not run on hardware
//
#include <hip/hip_runtime.h>

typedef __attribute__((ext_vector_type(16))) __bf16 v16bf;
typedef __attribute__((ext_vector_type(8)))  float  v8f;

union FragBF {
    v16bf v;
    uint4 q[2];
    __bf16 h[16];
};

// ---------------- async LDS staging (gfx1250) ----------------
#if defined(__AMDGCN__) && __has_builtin(__builtin_amdgcn_global_load_async_to_lds_b128)
#define HAVE_ASYNC_LDS 1
typedef int v4i_vs __attribute__((__vector_size__(16)));
typedef __attribute__((address_space(1))) v4i_vs* gas1_t;
typedef __attribute__((address_space(3))) v4i_vs* las3_t;
__device__ __forceinline__ void async_copy16(const void* g, void* l) {
    __builtin_amdgcn_global_load_async_to_lds_b128((gas1_t)g, (las3_t)l, 0, 0);
}
#if __has_builtin(__builtin_amdgcn_s_wait_asynccnt)
__device__ __forceinline__ void wait_async0() { __builtin_amdgcn_s_wait_asynccnt(0); }
#else
__device__ __forceinline__ void wait_async0() { asm volatile("s_wait_asynccnt 0" ::: "memory"); }
#endif
#else
#define HAVE_ASYNC_LDS 0
#endif

// ---------------- constants ----------------
constexpr long LTOT = 16384;        // h*w
constexpr int  CIN  = 256;
constexpr int  C2   = 512;
constexpr int  B    = 2;
constexpr float INV_S = 1.0f / 16384.0f;

// LDS strides (padded: 8-elem fragment groups stay 16B aligned, banks spread)
constexpr int LDA = 40;             // for 32-wide K tiles
constexpr int LDB = 136;            // for 128-wide N tiles
constexpr int LDBQ = 264;           // for 256-wide N tiles (qv kernel)

// ---------------- workspace layout (bytes) ----------------
constexpr size_t SZ_FULL   = (size_t)B * C2 * LTOT * 2;      // 33,554,432
constexpr size_t OFF_QFULL = 0;
constexpr size_t OFF_KFULL = OFF_QFULL + SZ_FULL;
constexpr size_t OFF_VFULL = OFF_KFULL + SZ_FULL;
constexpr size_t OFF_QV    = OFF_VFULL + SZ_FULL;
constexpr size_t OFF_Y     = OFF_QV + SZ_FULL;               // b*256*L bf16
constexpr size_t OFF_WPW   = OFF_Y + (size_t)B * CIN * LTOT * 2;
constexpr size_t OFF_WPROJ = OFF_WPW + (size_t)CIN * CIN * 2;
constexpr size_t OFF_KV    = OFF_WPROJ + (size_t)CIN * C2 * 2;
constexpr size_t OFF_KSUM  = OFF_KV + (size_t)B * 16 * 32 * 32 * 4;

// =====================================================================
// Elementwise: identity branch expand (fp32 -> bf16 into xfull[b][0..256))
// MODE 0: relu, MODE 1: scale
// =====================================================================
template <int MODE>
__global__ __launch_bounds__(256) void cvt_expand_kernel(
    const float* __restrict__ in, __bf16* __restrict__ out, float scale)
{
    long i  = (long)blockIdx.x * 256 + threadIdx.x;  // n = B*256*L
    long bb = i >> 22;                               // / (256*16384)
    float v = in[i];
    if (MODE == 0) v = v > 0.f ? v : 0.f;
    else           v *= scale;
    out[i + (bb << 22)] = (__bf16)v;                 // (bb*512 + c)*L + l
}

// =====================================================================
// Weight convert fp32 -> bf16
// =====================================================================
__global__ __launch_bounds__(256) void cvt_w_kernel(
    const float* __restrict__ in, __bf16* __restrict__ out, int n)
{
    int i = blockIdx.x * 256 + threadIdx.x;
    if (i < n) out[i] = (__bf16)in[i];
}

// =====================================================================
// Depthwise 5x5 "same" conv, groups=C.  x:[b*256,128,128] f32 -> y bf16
// =====================================================================
__global__ __launch_bounds__(256) void dwconv_kernel(
    const float* __restrict__ x, const float* __restrict__ wdw,
    __bf16* __restrict__ y)
{
    long gid = (long)blockIdx.x * 256 + threadIdx.x; // over B*256*L
    int  p   = (int)(gid & (LTOT - 1));
    long bc  = gid >> 14;
    int  c   = (int)(bc & 255);
    int  py  = p >> 7, px = p & 127;
    const float* xp = x + bc * LTOT;
    const float* wp = wdw + c * 25;
    float s = 0.f;
#pragma unroll
    for (int dy = -2; dy <= 2; ++dy) {
        int yy = py + dy;
        if ((unsigned)yy >= 128u) continue;
#pragma unroll
        for (int dx = -2; dx <= 2; ++dx) {
            int xx = px + dx;
            if ((unsigned)xx >= 128u) continue;
            s += wp[(dy + 2) * 5 + (dx + 2)] * xp[yy * 128 + xx];
        }
    }
    y[gid] = (__bf16)s;
}

// =====================================================================
// Tiled WMMA GEMM:  Out[256 x L] = A[256 x KDIM] * B[KDIM x L]  (per batch z)
// Block tile 64(M) x 128(N), 8 waves: waveM = wid&3, waveN = wid>>2.
// MODE 0: relu -> bf16 ; MODE 1: *scale -> bf16 ; MODE 2: -> f32
// =====================================================================
template <int KDIM, int MODE>
__global__ __launch_bounds__(256) void gemm_wmma_kernel(
    const __bf16* __restrict__ A, const __bf16* __restrict__ Bmat,
    void* __restrict__ Out, long outBatchStride, float scale)
{
    __shared__ __bf16 sA[64 * LDA];
    __shared__ __bf16 sB[32 * LDB];

    const int tid = threadIdx.x;
    const int lane = tid & 31, wid = tid >> 5;
    const int lh = lane >> 4, r16 = lane & 15;
    const int waveM = wid & 3, waveN = wid >> 2;
    const int bN = blockIdx.x, bM = blockIdx.y, bb = blockIdx.z;

    const __bf16* Bb = Bmat + (long)bb * KDIM * LTOT + (long)bN * 128;

    // staging coordinates
    const int ar = tid >> 2, ac = (tid & 3) << 3;       // A: 64x32
    const int br = tid >> 3, bc = (tid & 7) << 4;       // B: 32x128

    v8f acc[4] = {};

    for (int kt = 0; kt < KDIM; kt += 32) {
        __syncthreads();
#if HAVE_ASYNC_LDS
        async_copy16(A + (long)(bM * 64 + ar) * KDIM + kt + ac, &sA[ar * LDA + ac]);
        async_copy16(Bb + (long)(kt + br) * LTOT + bc,       &sB[br * LDB + bc]);
        async_copy16(Bb + (long)(kt + br) * LTOT + bc + 8,   &sB[br * LDB + bc + 8]);
        wait_async0();
#else
        *(uint4*)&sA[ar * LDA + ac] =
            *(const uint4*)(A + (long)(bM * 64 + ar) * KDIM + kt + ac);
        {
            const uint4* src = (const uint4*)(Bb + (long)(kt + br) * LTOT + bc);
            uint4* dst = (uint4*)&sB[br * LDB + bc];
            dst[0] = src[0];
            dst[1] = src[1];
        }
#endif
        __syncthreads();

        FragBF a;
        const __bf16* ap = &sA[(waveM * 16 + r16) * LDA + lh * 16];
        a.q[0] = *(const uint4*)ap;
        a.q[1] = *(const uint4*)(ap + 8);

#pragma unroll
        for (int nt = 0; nt < 4; ++nt) {
            int ncol = waveN * 64 + nt * 16 + r16;
            FragBF bf;
#pragma unroll
            for (int j = 0; j < 16; ++j)
                bf.h[j] = sB[(lh * 16 + j) * LDB + ncol];
            acc[nt] = __builtin_amdgcn_wmma_f32_16x16x32_bf16(
                false, a.v, false, bf.v, (short)0, acc[nt], false, false);
        }
    }

#pragma unroll
    for (int nt = 0; nt < 4; ++nt) {
        long col = (long)bN * 128 + waveN * 64 + nt * 16 + r16;
        int rowbase = bM * 64 + waveM * 16 + lh * 8;
#pragma unroll
        for (int i = 0; i < 8; ++i) {
            float v = acc[nt][i];
            long idx = (long)bb * outBatchStride + (long)(rowbase + i) * LTOT + col;
            if (MODE == 0) {
                v = v > 0.f ? v : 0.f;
                ((__bf16*)Out)[idx] = (__bf16)v;
            } else if (MODE == 1) {
                ((__bf16*)Out)[idx] = (__bf16)(v * scale);
            } else {
                ((float*)Out)[idx] = v;
            }
        }
    }
}

// =====================================================================
// kv[d,e] = sum_s K[d,s] * V[e,s] per (b,head).
// ksum computed on the matrix unit: ksum = K x ones, run UNCONDITIONALLY in
// every wave so EXEC stays all-ones around all WMMAs (ISA requirement);
// only the e0==0 waves store the result.
// Both fragments are direct contiguous 16B global loads (no LDS needed).
// =====================================================================
__global__ __launch_bounds__(128) void kv_kernel(
    const __bf16* __restrict__ kfull, const __bf16* __restrict__ vfull,
    float* __restrict__ kvbuf, float* __restrict__ ksumbuf)
{
    const int bh = blockIdx.x;                 // 0..31
    const int tid = threadIdx.x, lane = tid & 31, wid = tid >> 5;
    const int lh = lane >> 4, r16 = lane & 15;
    const int d0 = (wid >> 1) * 16, e0 = (wid & 1) * 16;
    const int bbt = bh >> 4, hh = bh & 15;

    const __bf16* kr = kfull + ((long)bbt * C2 + hh * 32 + d0 + r16) * LTOT;
    const __bf16* vr = vfull + ((long)bbt * C2 + hh * 32 + e0 + r16) * LTOT;

    FragBF ones;
#pragma unroll
    for (int j = 0; j < 16; ++j) ones.h[j] = (__bf16)1.0f;

    v8f acc  = {};
    v8f accS = {};

    for (int s0 = 0; s0 < LTOT; s0 += 32) {
        FragBF a, b;
        a.q[0] = *(const uint4*)(kr + s0 + lh * 8);
        a.q[1] = *(const uint4*)(kr + s0 + 16 + lh * 8);
        b.q[0] = *(const uint4*)(vr + s0 + lh * 8);
        b.q[1] = *(const uint4*)(vr + s0 + 16 + lh * 8);
        acc = __builtin_amdgcn_wmma_f32_16x16x32_bf16(
            false, a.v, false, b.v, (short)0, acc, false, false);
        accS = __builtin_amdgcn_wmma_f32_16x16x32_bf16(
            false, a.v, false, ones.v, (short)0, accS, false, false);
    }

#pragma unroll
    for (int i = 0; i < 8; ++i)
        kvbuf[((long)bh * 32 + d0 + lh * 8 + i) * 32 + e0 + r16] = acc[i];

    if (e0 == 0 && r16 == 0) {                 // every column of accS == ksum
#pragma unroll
        for (int i = 0; i < 8; ++i)
            ksumbuf[bh * 32 + d0 + lh * 8 + i] = accS[i] * INV_S;
    }
}

// =====================================================================
// qv[e,l] = z[l] * sum_d kv[d,e] * Q[d,l] per (b,head); 256-col tiles.
// z computed in-block from the staged Q tile and scaled ksum.
// =====================================================================
__global__ __launch_bounds__(256) void qv_kernel(
    const __bf16* __restrict__ qfull, const float* __restrict__ kvbuf,
    const float* __restrict__ ksumbuf, __bf16* __restrict__ qvfull)
{
    __shared__ __bf16 sA[32 * LDA];     // A[m=e][k=d] = kv[d][e] (transposed)
    __shared__ __bf16 sB[32 * LDBQ];    // B[d][n] = Q[d][l0+n]
    __shared__ float  sKs[32];
    __shared__ float  sZ[256];

    const int bh = blockIdx.y;          // 0..31
    const long l0 = (long)blockIdx.x * 256;
    const int bbt = bh >> 4, hh = bh & 15;
    const int tid = threadIdx.x, lane = tid & 31, wid = tid >> 5;
    const int lh = lane >> 4, r16 = lane & 15;
    const int waveM = wid >> 2, waveN = wid & 3;

    const __bf16* Qh = qfull + ((long)bbt * C2 + hh * 32) * LTOT + l0;

    {   // stage 32x256 Q tile (4x16B per thread)
        int r = tid >> 3, c = (tid & 7) << 5;
#if HAVE_ASYNC_LDS
#pragma unroll
        for (int i = 0; i < 4; ++i)
            async_copy16(Qh + (long)r * LTOT + c + i * 8, &sB[r * LDBQ + c + i * 8]);
#else
        const uint4* src = (const uint4*)(Qh + (long)r * LTOT + c);
        uint4* dst = (uint4*)&sB[r * LDBQ + c];
        dst[0] = src[0]; dst[1] = src[1]; dst[2] = src[2]; dst[3] = src[3];
#endif
    }
    // stage transposed kv tile (1024 elems) + ksum
#pragma unroll
    for (int i = 0; i < 4; ++i) {
        int idx = tid + i * 256;
        int d = idx >> 5, e = idx & 31;
        sA[e * LDA + d] = (__bf16)kvbuf[((long)bh * 32 + d) * 32 + e];
    }
    if (tid < 32) sKs[tid] = ksumbuf[bh * 32 + tid];
#if HAVE_ASYNC_LDS
    wait_async0();
#endif
    __syncthreads();

    {   // per-column z = 1/max(q . ksum_scaled, eps)
        float s = 0.f;
#pragma unroll
        for (int d = 0; d < 32; ++d)
            s += (float)sB[d * LDBQ + tid] * sKs[d];
        sZ[tid] = 1.0f / fmaxf(s, 1e-6f);
    }
    __syncthreads();

    FragBF a;
    const __bf16* ap = &sA[(waveM * 16 + r16) * LDA + lh * 16];
    a.q[0] = *(const uint4*)ap;
    a.q[1] = *(const uint4*)(ap + 8);

    v8f acc[4] = {};
#pragma unroll
    for (int nt = 0; nt < 4; ++nt) {
        int ncol = waveN * 64 + nt * 16 + r16;
        FragBF bf;
#pragma unroll
        for (int j = 0; j < 16; ++j)
            bf.h[j] = sB[(lh * 16 + j) * LDBQ + ncol];
        acc[nt] = __builtin_amdgcn_wmma_f32_16x16x32_bf16(
            false, a.v, false, bf.v, (short)0, acc[nt], false, false);
    }

#pragma unroll
    for (int nt = 0; nt < 4; ++nt) {
        int ncol = waveN * 64 + nt * 16 + r16;
        int m0 = waveM * 16 + lh * 8;
#pragma unroll
        for (int i = 0; i < 8; ++i) {
            float v = acc[nt][i] * sZ[ncol];
            qvfull[((long)bbt * C2 + hh * 32 + m0 + i) * LTOT + l0 + ncol] = (__bf16)v;
        }
    }
}

// =====================================================================
// Host launch
// =====================================================================
extern "C" void kernel_launch(void* const* d_in, const int* in_sizes, int n_in,
                              void* d_out, int out_size, void* d_ws, size_t ws_size,
                              hipStream_t stream)
{
    (void)in_sizes; (void)n_in; (void)out_size; (void)ws_size;

    const float* xin[3]  = {(const float*)d_in[0], (const float*)d_in[1], (const float*)d_in[2]};
    const float* wdw[3]  = {(const float*)d_in[3], (const float*)d_in[5], (const float*)d_in[7]};
    const float* wpw[3]  = {(const float*)d_in[4], (const float*)d_in[6], (const float*)d_in[8]};
    const float* w_proj  = (const float*)d_in[9];

    char* ws = (char*)d_ws;
    __bf16* qfull  = (__bf16*)(ws + OFF_QFULL);
    __bf16* kfull  = (__bf16*)(ws + OFF_KFULL);
    __bf16* vfull  = (__bf16*)(ws + OFF_VFULL);
    __bf16* qvfull = (__bf16*)(ws + OFF_QV);
    __bf16* ybuf   = (__bf16*)(ws + OFF_Y);
    __bf16* wbf    = (__bf16*)(ws + OFF_WPW);
    __bf16* wprojbf= (__bf16*)(ws + OFF_WPROJ);
    float*  kvbuf  = (float*)(ws + OFF_KV);
    float*  ksumbuf= (float*)(ws + OFF_KSUM);

    __bf16* xfull[3] = {qfull, kfull, vfull};

    const dim3 gGemm(128, 4, B);                // L/128, 256/64, batch
    const int gElem = (B * CIN * (int)LTOT) / 256;  // 32,768

    for (int t = 0; t < 3; ++t) {
        if (t < 2)
            cvt_expand_kernel<0><<<gElem, 256, 0, stream>>>(xin[t], xfull[t], 1.0f);
        else
            cvt_expand_kernel<1><<<gElem, 256, 0, stream>>>(xin[t], xfull[t], INV_S);
        dwconv_kernel<<<gElem, 256, 0, stream>>>(xin[t], wdw[t], ybuf);
        cvt_w_kernel<<<256, 256, 0, stream>>>(wpw[t], wbf, CIN * CIN);
        if (t < 2)
            gemm_wmma_kernel<256, 0><<<gGemm, 256, 0, stream>>>(
                wbf, ybuf, xfull[t] + (long)CIN * LTOT, (long)C2 * LTOT, 1.0f);
        else
            gemm_wmma_kernel<256, 1><<<gGemm, 256, 0, stream>>>(
                wbf, ybuf, xfull[t] + (long)CIN * LTOT, (long)C2 * LTOT, INV_S);
    }

    kv_kernel<<<B * 16, 128, 0, stream>>>(kfull, vfull, kvbuf, ksumbuf);
    qv_kernel<<<dim3(64, B * 16), 256, 0, stream>>>(qfull, kvbuf, ksumbuf, qvfull);

    cvt_w_kernel<<<512, 256, 0, stream>>>(w_proj, wprojbf, CIN * C2);
    gemm_wmma_kernel<512, 2><<<gGemm, 256, 0, stream>>>(
        wprojbf, qvfull, d_out, (long)CIN * LTOT, 1.0f);
}
